// Multimodal_29222957482897
// MI455X (gfx1250) — matrix operand order repaired and checked by reading, not compiled, not run on hardware
//
#include <hip/hip_runtime.h>

// ---------------------------------------------------------------------------
// MCAT-style multimodal forward (WSI LSH self-attn + omic cross-attn) on
// gfx1250. All matmuls go through v_wmma_f32_16x16x32_f16 (wave32 WMMA).
// WMMA operands are staged in LDS pair-packed (2 x f16 per dword) along the
// contraction dim so operand fetches are ds_load_b32, not ds_load_u16.
// ---------------------------------------------------------------------------

typedef __attribute__((ext_vector_type(16))) _Float16 v16h;
typedef __attribute__((ext_vector_type(8)))  float    v8f;
typedef __attribute__((ext_vector_type(8)))  unsigned v8u;

#define T_PAD   10240      // padded sequence (10000 + 240)
#define NPATH   10000
#define DMODEL  256
#define DH      64
#define NHEADS  4
#define NHASH   2
#define NBKT    160        // nh * n_buckets = 2*80
#define NCHUNK  160        // chunks per head-batch
#define NSORT   20480      // NHASH * T_PAD

__device__ inline unsigned pack2(float lo, float hi) {
  union { _Float16 h[2]; unsigned u; } p;
  p.h[0] = (_Float16)lo;
  p.h[1] = (_Float16)hi;
  return p.u;
}

// ------------------------------ WMMA helpers -------------------------------
// 16x32 f16 A tile from dword-packed LDS ([16 rows][>=16 dwords], pair=(k,k+1)).
// Lane mapping per CDNA5 ISA 7.12.2 (wave32):
//   lanes 0-15:  M=lane,   K in {0..7, 16..23}  -> pair idx {0..3, 8..11}
//   lanes 16-31: M=lane-16,K in {8..15, 24..31} -> pair idx {4..7, 12..15}
__device__ inline v16h load_a_tile(const unsigned* s, int ldu, int lane) {
  int m = lane & 15, grp = lane >> 4;
  v8u t;
#pragma unroll
  for (int v = 0; v < 8; ++v) {
    int k2 = ((v < 4) ? v : (8 + (v - 4))) + 4 * grp;
    t[v] = s[m * ldu + k2];
  }
  return __builtin_bit_cast(v16h, t);
}
// 32x16 f16 B tile from dword-packed LDS ([16 k-pairs][N dwords-of-columns]):
// element dword at [k2][n] holds (K=2*k2, K=2*k2+1) for column n.
__device__ inline v16h load_b_tile(const unsigned* s, int ldu, int lane) {
  int n = lane & 15, grp = lane >> 4;
  v8u t;
#pragma unroll
  for (int v = 0; v < 8; ++v) {
    int k2 = ((v < 4) ? v : (8 + (v - 4))) + 4 * grp;
    t[v] = s[k2 * ldu + n];
  }
  return __builtin_bit_cast(v16h, t);
}
__device__ inline v8f wmma16(v16h a, v16h b, v8f c) {
  return __builtin_amdgcn_wmma_f32_16x16x32_f16(false, a, false, b, (short)0, c,
                                                false, false);
}

// ------------------------------ generic GEMM -------------------------------
// C[M,256] = act(A[M,K] @ W[K,256] + bias). Rows >= arows of A read as zero
// (implements the zero-padded xs rows for the LSH projections).
// Block: 256 thr = 8 waves; tile 128x64; K-step 32.
template <int ACT>
__global__ __launch_bounds__(256) void k_gemm(const float* __restrict__ A,
                                              int arows, int M, int K,
                                              const float* __restrict__ W,
                                              const float* __restrict__ bias,
                                              float* __restrict__ C, int N) {
  __shared__ unsigned As[128][20];   // 16 k-pairs + 4 pad dwords
  __shared__ unsigned Bs[16][72];    // 16 k-pairs x (64 cols + 8 pad)
  int tid = threadIdx.x, lane = tid & 31, wave = tid >> 5;
  int rowBase = blockIdx.y * 128;
  int colBase = blockIdx.x * 64;
  v8f z = {};
  v8f acc[4];
#pragma unroll
  for (int i = 0; i < 4; ++i) acc[i] = z;

  for (int k0 = 0; k0 < K; k0 += 32) {
    for (int i = tid; i < 128 * 16; i += 256) {
      int r = i >> 4, k2 = i & 15;
      int gr = rowBase + r;
      unsigned u = 0u;
      if (gr < arows) {
        const float* ap = A + (size_t)gr * K + k0 + 2 * k2;
        u = pack2(ap[0], ap[1]);
      }
      As[r][k2] = u;
    }
    for (int i = tid; i < 16 * 64; i += 256) {
      int k2 = i >> 6, n = i & 63;
      const float* wp = W + (size_t)(k0 + 2 * k2) * N + colBase + n;
      Bs[k2][n] = pack2(wp[0], wp[N]);
    }
    __syncthreads();
    v16h a = load_a_tile(&As[wave * 16][0], 20, lane);
#pragma unroll
    for (int nt = 0; nt < 4; ++nt) {
      v16h b = load_b_tile(&Bs[0][nt * 16], 72, lane);
      acc[nt] = wmma16(a, b, acc[nt]);
    }
    __syncthreads();
  }
  int grp = lane >> 4, n0 = lane & 15;
#pragma unroll
  for (int nt = 0; nt < 4; ++nt) {
    int col = colBase + nt * 16 + n0;
#pragma unroll
    for (int r = 0; r < 8; ++r) {
      int row = rowBase + wave * 16 + r + 8 * grp;
      if (row < M) {
        float v = acc[nt][r];
        if (bias) v += bias[col];
        if (ACT == 1) v = fmaxf(v, 0.f);
        else if (ACT == 2) v = tanhf(v);
        else if (ACT == 3) v = 1.f / (1.f + expf(-v));
        C[(size_t)row * N + col] = v;
      }
    }
  }
}

// ------------------------------ SNN (omics) --------------------------------
__global__ __launch_bounds__(256) void k_snn(const float* __restrict__ x, int sz,
                                             const float* __restrict__ w1,
                                             const float* __restrict__ b1,
                                             const float* __restrict__ w2,
                                             const float* __restrict__ b2,
                                             float* __restrict__ out) {
  __shared__ float sx[608];
  __shared__ float sh[256];
  int tid = threadIdx.x;
  for (int i = tid; i < sz; i += 256) sx[i] = x[i];
  __syncthreads();
  float a = b1[tid];
  for (int i = 0; i < sz; ++i) a += sx[i] * w1[i * 256 + tid];
  a = (a > 0.f) ? a : (expf(a) - 1.f);   // ELU
  sh[tid] = a;
  __syncthreads();
  float a2 = b2[tid];
  for (int i = 0; i < 256; ++i) a2 += sh[i] * w2[i * 256 + tid];
  a2 = (a2 > 0.f) ? a2 : (expf(a2) - 1.f);
  out[tid] = a2;
}

// ------------------------- LSH bucketing + sort ----------------------------
// rot: [64][2][40]; qk: [T][256] (head h occupies cols h*64..). buckets[b][h*T+t]
__global__ void k_lsh_buckets(const float* __restrict__ qk,
                              const float* __restrict__ rot,
                              int* __restrict__ buckets) {
  int idx = blockIdx.x * blockDim.x + threadIdx.x;
  if (idx >= NHEADS * T_PAD) return;
  int b = idx / T_PAD, t = idx % T_PAD;
  const float* q = qk + (size_t)t * DMODEL + b * DH;
  for (int h = 0; h < NHASH; ++h) {
    float s[40];
#pragma unroll
    for (int i = 0; i < 40; ++i) s[i] = 0.f;
    for (int f = 0; f < DH; ++f) {
      float qv = q[f];
      const float* rp = rot + f * 80 + h * 40;
#pragma unroll
      for (int i = 0; i < 40; ++i) s[i] += qv * rp[i];
    }
    float best = s[0];
    int bi = 0;
    for (int i = 1; i < 80; ++i) {
      float v = (i < 40) ? s[i] : -s[i - 40];
      if (v > best) { best = v; bi = i; }
    }
    buckets[b * NSORT + h * T_PAD + t] = bi + h * 80;
  }
}

__global__ __launch_bounds__(256) void k_hist(const int* __restrict__ buckets,
                                              int* __restrict__ bases) {
  __shared__ int cnt[NBKT];
  int b = blockIdx.x;
  for (int i = threadIdx.x; i < NBKT; i += blockDim.x) cnt[i] = 0;
  __syncthreads();
  for (int i = threadIdx.x; i < NSORT; i += blockDim.x)
    atomicAdd(&cnt[buckets[b * NSORT + i]], 1);
  __syncthreads();
  if (threadIdx.x == 0) {
    int run = 0;
    for (int k = 0; k < NBKT; ++k) { bases[b * NBKT + k] = run; run += cnt[k]; }
  }
}

// Stable counting-sort scatter: one wave per (b, bucket); ballot-based
// intra-wave stable ranking. key = T*bucket + i%T  =>  (bucket asc, i asc).
__global__ __launch_bounds__(256) void k_scatter(const int* __restrict__ buckets,
                                                 const int* __restrict__ bases,
                                                 int* __restrict__ sticker) {
  int w = blockIdx.x * 8 + (threadIdx.x >> 5);
  int lane = threadIdx.x & 31;
  int b = w / NBKT, bk = w % NBKT;
  const int* B = buckets + b * NSORT;
  int* S = sticker + b * NSORT;
  int base = bases[b * NBKT + bk];
  int cnt = 0;
  for (int i0 = 0; i0 < NSORT; i0 += 32) {
    int i = i0 + lane;
    bool match = (B[i] == bk);
    unsigned m = (unsigned)__ballot(match);
    unsigned before = m & ((1u << lane) - 1u);
    if (match) S[base + cnt + __popc(before)] = i;
    cnt += __popc(m);
  }
}

// -------------------------- LSH chunk attention ----------------------------
// One 128-row chunk per block (8 waves x 16 rows). Flash-style online softmax
// over 256 look-back keys in 64-key LDS tiles; QK^T and PV on WMMA.
__global__ __launch_bounds__(256) void k_lsh_chunk(
    const float* __restrict__ qk, const float* __restrict__ vv,
    const int* __restrict__ sticker, float* __restrict__ soh,
    float* __restrict__ slog) {
  int b = blockIdx.x / NCHUNK;
  int c = blockIdx.x % NCHUNK;
  int cp = (c + NCHUNK - 1) % NCHUNK;
  int tid = threadIdx.x, lane = tid & 31, wave = tid >> 5;

  __shared__ unsigned Qs[128][36];     // A: 32 e-pairs + 4 pad
  __shared__ unsigned KTe[32][72];     // B for QK^T: [e-pair][j]
  __shared__ unsigned Vj[32][72];      // B for PV:   [j-pair][e]
  __shared__ unsigned Ps[8][16][20];   // A: per-wave P, 16 j-pairs + 4 pad
  __shared__ int   QT[128];
  __shared__ int   KT[64];
  __shared__ float RN[64];
  __shared__ float Mrow[8][16], Lrow[8][16];

  const int* stk = sticker + b * NSORT;

  for (int i = tid; i < 128; i += 256) QT[i] = stk[c * 128 + i] % T_PAD;
  __syncthreads();
  for (int i = tid; i < 128 * 32; i += 256) {
    int r = i >> 5, e2 = i & 31;
    const float* qp = qk + (size_t)QT[r] * DMODEL + b * DH + 2 * e2;
    Qs[r][e2] = pack2(qp[0], qp[1]);
  }
  if (tid < 128) { Mrow[tid >> 4][tid & 15] = -__builtin_inff();
                   Lrow[tid >> 4][tid & 15] = 0.f; }
  __syncthreads();

  v16h aQ0 = load_a_tile(&Qs[wave * 16][0], 36, lane);   // e 0..31
  v16h aQ1 = load_a_tile(&Qs[wave * 16][16], 36, lane);  // e 32..63
  v8f z = {};
  v8f O[4];
#pragma unroll
  for (int i = 0; i < 4; ++i) O[i] = z;

  const float scale = 0.125f;   // DH^-0.5
  int grp = lane >> 4, n0 = lane & 15;

  for (int jt = 0; jt < 4; ++jt) {
    __syncthreads();   // prior tile fully consumed
    for (int jj = tid; jj < 64; jj += 256) {
      int j = jt * 64 + jj;
      int srcChunk = (j < 128) ? c : cp;   // look_back: keys from chunk c, c-1
      KT[jj] = stk[srcChunk * 128 + (j & 127)] % T_PAD;
    }
    __syncthreads();
    if (tid < 64) {
      int t = KT[tid];
      float ss = 0.f;
      for (int e = 0; e < DH; ++e) {
        float v = qk[(size_t)t * DMODEL + b * DH + e];
        ss += v * v;
      }
      RN[tid] = 1.f / fmaxf(sqrtf(ss), 1e-6f);
    }
    __syncthreads();
    for (int i = tid; i < 32 * 64; i += 256) {
      int p = i >> 6, x = i & 63;
      // KTe[e-pair p][column j=x]: normalized keys, transposed
      {
        int t = KT[x];
        float rn = RN[x];
        const float* kp = qk + (size_t)t * DMODEL + b * DH + 2 * p;
        KTe[p][x] = pack2(kp[0] * rn, kp[1] * rn);
      }
      // Vj[j-pair p][column e=x]: values, pair = two consecutive keys
      {
        int t0 = KT[2 * p], t1 = KT[2 * p + 1];
        Vj[p][x] = pack2(vv[(size_t)t0 * DMODEL + b * DH + x],
                         vv[(size_t)t1 * DMODEL + b * DH + x]);
      }
    }
    __syncthreads();

    for (int kh = 0; kh < 2; ++kh) {
      v8f S[2];
#pragma unroll
      for (int sub = 0; sub < 2; ++sub) {
        v8f s = z;
        v16h b0 = load_b_tile(&KTe[0][kh * 32 + sub * 16], 72, lane);
        v16h b1 = load_b_tile(&KTe[16][kh * 32 + sub * 16], 72, lane);
        s = wmma16(aQ0, b0, s);
        s = wmma16(aQ1, b1, s);
        S[sub] = s;
      }
      // scale + padding mask + self-attention mask
#pragma unroll
      for (int r = 0; r < 8; ++r) {
        int m = r + 8 * grp;
        int qt = QT[wave * 16 + m];
        bool qv = qt < NPATH;
#pragma unroll
        for (int sub = 0; sub < 2; ++sub) {
          int jl = kh * 32 + sub * 16 + n0;
          int kt = KT[jl];
          float sval = S[sub][r] * scale;
          if (!(qv && (kt < NPATH))) sval = -1e9f;
          if (qt == kt) sval = -5e4f;
          S[sub][r] = sval;
        }
      }
      // online softmax per row (rows live in 16-lane groups)
      float corrv[8];
#pragma unroll
      for (int r = 0; r < 8; ++r) {
        int m = r + 8 * grp;
        float rmax = fmaxf(S[0][r], S[1][r]);
#pragma unroll
        for (int off = 1; off < 16; off <<= 1)
          rmax = fmaxf(rmax, __shfl_xor(rmax, off, 32));
        float mold = Mrow[wave][m];
        float mnew = fmaxf(mold, rmax);
        float corr = expf(mold - mnew);
        float p0 = expf(S[0][r] - mnew);
        float p1 = expf(S[1][r] - mnew);
        float rsum = p0 + p1;
#pragma unroll
        for (int off = 1; off < 16; off <<= 1)
          rsum += __shfl_xor(rsum, off, 32);
        if (n0 == 0) {
          Mrow[wave][m] = mnew;
          Lrow[wave][m] = Lrow[wave][m] * corr + rsum;
        }
        corrv[r] = corr;
        _Float16* ph = (_Float16*)&Ps[wave][m][0];
        ph[n0]      = (_Float16)p0;
        ph[16 + n0] = (_Float16)p1;
      }
      __syncthreads();   // uniform; makes P visible for the A-operand load
#pragma unroll
      for (int nt = 0; nt < 4; ++nt)
#pragma unroll
        for (int r = 0; r < 8; ++r) O[nt][r] *= corrv[r];
      v16h aP = load_a_tile(&Ps[wave][0][0], 20, lane);
#pragma unroll
      for (int nt = 0; nt < 4; ++nt) {
        v16h bV = load_b_tile(&Vj[kh * 16][nt * 16], 72, lane);
        O[nt] = wmma16(aP, bV, O[nt]);
      }
    }
  }
  __syncthreads();
  // scatter directly to un-sorted (hash, token) positions: undo == inverse map
#pragma unroll
  for (int r = 0; r < 8; ++r) {
    int m = r + 8 * grp;
    int s = stk[c * 128 + wave * 16 + m];
    int h = s / T_PAD, t = s % T_PAD;
    float l = Lrow[wave][m];
    float inv = 1.f / l;
#pragma unroll
    for (int nt = 0; nt < 4; ++nt)
      soh[(((size_t)(b * NHASH + h)) * T_PAD + t) * DH + nt * 16 + n0] =
          O[nt][r] * inv;
    if (n0 == 0)
      slog[((size_t)(b * NHASH + h)) * T_PAD + t] = Mrow[wave][m] + logf(l);
  }
}

// combine the two hash rounds:  o = sum_h o_h * softmax_h(lse_h)
__global__ void k_lsh_combine(const float* __restrict__ soh,
                              const float* __restrict__ slog,
                              float* __restrict__ att) {
  int idx = blockIdx.x * blockDim.x + threadIdx.x;
  if (idx >= NHEADS * T_PAD * DH) return;
  int e = idx & 63;
  int bt = idx >> 6;
  int b = bt / T_PAD, t = bt % T_PAD;
  float l0 = slog[(size_t)(b * NHASH) * T_PAD + t];
  float l1 = slog[(size_t)(b * NHASH + 1) * T_PAD + t];
  float mm = fmaxf(l0, l1);
  float w0 = expf(l0 - mm), w1 = expf(l1 - mm);
  float zz = w0 + w1;
  float o = (w0 * soh[((size_t)(b * NHASH) * T_PAD + t) * DH + e] +
             w1 * soh[((size_t)(b * NHASH + 1) * T_PAD + t) * DH + e]) / zz;
  att[(size_t)t * DMODEL + b * DH + e] = o;
}

// ------------------------- residual + layernorm ----------------------------
__global__ __launch_bounds__(256) void k_add_ln(const float* __restrict__ x,
                                                const float* __restrict__ res,
                                                const float* __restrict__ g,
                                                const float* __restrict__ bta,
                                                float* __restrict__ y) {
  int row = blockIdx.x, c = threadIdx.x;
  __shared__ float red[256];
  float v = x[(size_t)row * 256 + c] + res[(size_t)row * 256 + c];
  red[c] = v;
  __syncthreads();
  for (int off = 128; off > 0; off >>= 1) {
    if (c < off) red[c] += red[c + off];
    __syncthreads();
  }
  float mu = red[0] * (1.f / 256.f);
  __syncthreads();
  float d = v - mu;
  red[c] = d * d;
  __syncthreads();
  for (int off = 128; off > 0; off >>= 1) {
    if (c < off) red[c] += red[c + off];
    __syncthreads();
  }
  float var = red[0] * (1.f / 256.f);
  y[(size_t)row * 256 + c] = d * rsqrtf(var + 1e-5f) * g[c] + bta[c];
}

// ------------------------- generic small MHA core --------------------------
// One block (64 thr) per (query, head); chunked 2-pass softmax over Lk keys.
__global__ __launch_bounds__(64) void k_attn(const float* __restrict__ qh,
                                             const float* __restrict__ kh,
                                             const float* __restrict__ vh,
                                             float* __restrict__ o, int Lk) {
  int q = blockIdx.x, h = blockIdx.y, tid = threadIdx.x;
  __shared__ float sQ[64], sP[64], red[64];
  sQ[tid] = qh[(size_t)q * DMODEL + h * DH + tid];
  __syncthreads();
  float lmax = -__builtin_inff();
  for (int k0 = 0; k0 < Lk; k0 += 64) {
    int k = k0 + tid;
    float s = -__builtin_inff();
    if (k < Lk) {
      s = 0.f;
      const float* kr = kh + (size_t)k * DMODEL + h * DH;
      for (int e = 0; e < DH; ++e) s += sQ[e] * kr[e];
      s *= 0.125f;
    }
    lmax = fmaxf(lmax, s);
  }
  red[tid] = lmax;
  __syncthreads();
  for (int off = 32; off > 0; off >>= 1) {
    if (tid < off) red[tid] = fmaxf(red[tid], red[tid + off]);
    __syncthreads();
  }
  float gmax = red[0];
  __syncthreads();
  float denom = 0.f, acc = 0.f;
  for (int k0 = 0; k0 < Lk; k0 += 64) {
    int k = k0 + tid;
    float p = 0.f;
    if (k < Lk) {
      float s = 0.f;
      const float* kr = kh + (size_t)k * DMODEL + h * DH;
      for (int e = 0; e < DH; ++e) s += sQ[e] * kr[e];
      p = expf(s * 0.125f - gmax);
    }
    sP[tid] = p;
    __syncthreads();
    int lim = min(64, Lk - k0);
    for (int kk = 0; kk < lim; ++kk) {
      float pv = sP[kk];
      denom += pv;
      acc += pv * vh[(size_t)(k0 + kk) * DMODEL + h * DH + tid];
    }
    __syncthreads();
  }
  o[(size_t)q * DMODEL + h * DH + tid] = acc / denom;
}

// ---------------------------- gated attention pool -------------------------
__global__ void k_gate_score(const float* __restrict__ a,
                             const float* __restrict__ b,
                             const float* __restrict__ cW,
                             const float* __restrict__ cb,
                             float* __restrict__ A, int N) {
  int n = blockIdx.x * blockDim.x + threadIdx.x;
  if (n >= N) return;
  float s = 0.f;
  for (int j = 0; j < 256; ++j)
    s += a[(size_t)n * 256 + j] * b[(size_t)n * 256 + j] * cW[j];
  A[n] = s + cb[0];
}

__global__ __launch_bounds__(256) void k_softmax_1d(const float* __restrict__ A,
                                                    float* __restrict__ w, int N) {
  __shared__ float red[256];
  __shared__ float sm, ss;
  int tid = threadIdx.x;
  float m = -__builtin_inff();
  for (int i = tid; i < N; i += 256) m = fmaxf(m, A[i]);
  red[tid] = m;
  __syncthreads();
  for (int off = 128; off > 0; off >>= 1) {
    if (tid < off) red[tid] = fmaxf(red[tid], red[tid + off]);
    __syncthreads();
  }
  if (tid == 0) sm = red[0];
  __syncthreads();
  float s = 0.f;
  for (int i = tid; i < N; i += 256) s += expf(A[i] - sm);
  red[tid] = s;
  __syncthreads();
  for (int off = 128; off > 0; off >>= 1) {
    if (tid < off) red[tid] += red[tid + off];
    __syncthreads();
  }
  if (tid == 0) ss = red[0];
  __syncthreads();
  for (int i = tid; i < N; i += 256) w[i] = expf(A[i] - sm) / ss;
}

__global__ __launch_bounds__(256) void k_weighted_sum(const float* __restrict__ x,
                                                      const float* __restrict__ w,
                                                      int N, float* __restrict__ out) {
  int col = blockIdx.x, tid = threadIdx.x;
  __shared__ float red[256];
  float s = 0.f;
  for (int n = tid; n < N; n += 256) s += w[n] * x[(size_t)n * 256 + col];
  red[tid] = s;
  __syncthreads();
  for (int off = 128; off > 0; off >>= 1) {
    if (tid < off) red[tid] += red[tid + off];
    __syncthreads();
  }
  if (tid == 0) out[col] = red[0];
}

template <int ACT>
__global__ __launch_bounds__(256) void k_vec_linear(const float* __restrict__ x,
                                                    const float* __restrict__ W,
                                                    const float* __restrict__ b,
                                                    float* __restrict__ y) {
  __shared__ float sx[256];
  int j = threadIdx.x;
  sx[j] = x[j];
  __syncthreads();
  float s = b[j];
  for (int i = 0; i < 256; ++i) s += sx[i] * W[i * 256 + j];
  if (ACT == 1) s = fmaxf(s, 0.f);
  y[j] = s;
}

// ---------------------------- final fusion head ----------------------------
__global__ __launch_bounds__(256) void k_final(
    const float* __restrict__ hvec,      // [4][256]: hps,hpc,hos,hoc
    const float* fqW, const float* fqb, const float* fkW, const float* fkb,
    const float* cW, const float* cb, const float* cpsW, const float* cpsb,
    const float* cpcW, const float* cpcb, const float* cosW, const float* cosb,
    const float* cocW, const float* cocb, float* __restrict__ out) {
  __shared__ float H[4][256], Q[4][256], Kk[4][256], Amat[16], colw[4], HF[256];
  int tid = threadIdx.x;
  for (int i = 0; i < 4; ++i) H[i][tid] = hvec[i * 256 + tid];
  __syncthreads();
  for (int i = 0; i < 4; ++i) {
    float sq = fqb[tid], sk = fkb[tid];
    for (int r = 0; r < 256; ++r) {
      sq += H[i][r] * fqW[r * 256 + tid];
      sk += H[i][r] * fkW[r * 256 + tid];
    }
    Q[i][tid] = sq;
    Kk[i][tid] = sk;
  }
  __syncthreads();
  if (tid < 16) {
    int i = tid >> 2, l = tid & 3;
    float s = 0.f;
    for (int r = 0; r < 256; ++r) s += Q[i][r] * Kk[l][r];
    Amat[tid] = s;
  }
  __syncthreads();
  if (tid < 4) {
    int i = tid;
    float m = Amat[i * 4];
    for (int l = 1; l < 4; ++l) m = fmaxf(m, Amat[i * 4 + l]);
    float zz = 0.f;
    for (int l = 0; l < 4; ++l) {
      float e = expf(Amat[i * 4 + l] - m);
      Amat[i * 4 + l] = e;
      zz += e;
    }
    for (int l = 0; l < 4; ++l) Amat[i * 4 + l] /= zz;
  }
  __syncthreads();
  if (tid < 4)
    colw[tid] = Amat[tid] + Amat[4 + tid] + Amat[8 + tid] + Amat[12 + tid];
  __syncthreads();
  float hf = 0.f;
  for (int l = 0; l < 4; ++l) hf += colw[l] * H[l][tid];
  HF[tid] = hf;
  __syncthreads();
  if (tid < 4) {
    int o = tid;
    float s = cb[o];
    for (int r = 0; r < 256; ++r) s += HF[r] * cW[r * 4 + o];
    out[o] = s;
    s = cpsb[o];
    for (int r = 0; r < 256; ++r) s += H[0][r] * cpsW[r * 4 + o];
    out[4 + o] = s;
    s = cpcb[o];
    for (int r = 0; r < 256; ++r) s += H[1][r] * cpcW[r * 4 + o];
    out[8 + o] = s;
    s = cosb[o];
    for (int r = 0; r < 256; ++r) s += H[2][r] * cosW[r * 4 + o];
    out[12 + o] = s;
    s = cocb[o];
    for (int r = 0; r < 256; ++r) s += H[3][r] * cocW[r * 4 + o];
    out[16 + o] = s;
  }
}

// ------------------------------- host side ---------------------------------
static inline void launch_gemm(hipStream_t st, const float* A, int arows, int M,
                               int K, const float* W, const float* bias,
                               float* C, int act) {
  dim3 grid(4, (M + 127) / 128);
  switch (act) {
    case 1: k_gemm<1><<<grid, 256, 0, st>>>(A, arows, M, K, W, bias, C, 256); break;
    case 2: k_gemm<2><<<grid, 256, 0, st>>>(A, arows, M, K, W, bias, C, 256); break;
    case 3: k_gemm<3><<<grid, 256, 0, st>>>(A, arows, M, K, W, bias, C, 256); break;
    default: k_gemm<0><<<grid, 256, 0, st>>>(A, arows, M, K, W, bias, C, 256); break;
  }
}

extern "C" void kernel_launch(void* const* d_in, const int* in_sizes, int n_in,
                              void* d_out, int out_size, void* d_ws,
                              size_t ws_size, hipStream_t stream) {
  (void)in_sizes; (void)n_in; (void)out_size; (void)ws_size;
  const float* x_path = (const float*)d_in[0];
  const float* P = (const float*)d_in[1];
  const float* xo[6];
  for (int i = 0; i < 6; ++i) xo[i] = (const float*)d_in[2 + i];
  float* out = (float*)d_out;

  // ---- params: flat buffer in JAX pytree order (dict keys sorted) ----
  size_t off = 0;
  auto take = [&](size_t n) { const float* p = P + off; off += n; return p; };
  const float *c_W = take(1024), *c_b = take(4);
  const float *coc_W = take(1024), *coc_b = take(4);
  const float *cos_W = take(1024), *cos_b = take(4);
  const float *cpc_W = take(1024), *cpc_b = take(4);
  const float *cps_W = take(1024), *cps_b = take(4);
  const float *fk_W = take(65536), *fk_b = take(256);
  const float *fq_W = take(65536), *fq_b = take(256);
  struct GateP { const float *aW, *ab, *bW, *bb, *cW, *cb; };
  auto takeGate = [&]() {
    GateP g;
    g.aW = take(65536); g.ab = take(256);
    g.bW = take(65536); g.bb = take(256);
    g.cW = take(256);   g.cb = take(1);
    return g;
  };
  GateP g_oc = takeGate(), g_os = takeGate(), g_pc = takeGate(), g_ps = takeGate();
  struct LnP { const float *b, *g; };
  auto takeLn = [&]() { LnP l; l.b = take(256); l.g = take(256); return l; };
  LnP ln_oc = takeLn(), ln_os = takeLn(), ln_pc = takeLn(), ln_ps = takeLn();
  const float *lsh_Wo = take(65536), *lsh_Wqk = take(65536),
              *lsh_Wv = take(65536), *lsh_bo = take(256);
  struct MhaP { const float *Wk, *Wo, *Wq, *Wv, *bk, *bo, *bq, *bv; };
  auto takeMha = [&]() {
    MhaP m;
    m.Wk = take(65536); m.Wo = take(65536); m.Wq = take(65536); m.Wv = take(65536);
    m.bk = take(256); m.bo = take(256); m.bq = take(256); m.bv = take(256);
    return m;
  };
  MhaP mha_oc = takeMha(), mha_os = takeMha(), mha_pc = takeMha();
  const float *roc_W = take(65536), *roc_b = take(256);
  const float *ros_W = take(65536), *ros_b = take(256);
  const float *rpc_W = take(65536), *rpc_b = take(256);
  const float *rps_W = take(65536), *rps_b = take(256);
  const float *rot = take(64 * 2 * 40);
  const int OSZ[6] = {100, 200, 300, 400, 500, 600};
  const float *sW1[6], *sB1[6], *sW2[6], *sB2[6];
  for (int i = 0; i < 6; ++i) {
    sW1[i] = take((size_t)OSZ[i] * 256); sB1[i] = take(256);
    sW2[i] = take(65536);                sB2[i] = take(256);
  }
  const float *wsi_W = take((size_t)1024 * 256), *wsi_b = take(256);

  // ---- workspace bump allocator (region reuse noted per phase) ----
  char* wp = (char*)d_ws;
  auto af = [&](size_t n) { float* p = (float*)wp; wp += n * sizeof(float); return p; };
  auto ai = [&](size_t n) { int* p = (int*)wp; wp += n * sizeof(int); return p; };
  float* hp   = af((size_t)NPATH * DMODEL);
  float* hob  = af(6 * DMODEL);
  float* hs   = af((size_t)NPATH * DMODEL);
  float* hc   = af((size_t)NPATH * DMODEL);
  float* hs2  = af(6 * DMODEL);
  float* hc2  = af(6 * DMODEL);
  float* q6   = af(6 * DMODEL);
  float* k6   = af(6 * DMODEL);
  float* v6   = af(6 * DMODEL);
  float* o6   = af(6 * DMODEL);
  float* pre6 = af(6 * DMODEL);
  float* pool_tmp = af(DMODEL);
  float* hvec = af(4 * DMODEL);            // hps,hpc,hos,hoc
  float* Ascore = af(T_PAD);
  float* gw     = af(T_PAD);
  float* slog   = af((size_t)NHEADS * NHASH * T_PAD);
  int* buckets  = ai((size_t)NHEADS * NSORT);
  int* sticker  = ai((size_t)NHEADS * NSORT);
  int* bases    = ai(NHEADS * NBKT);
  float* BIG1 = af((size_t)T_PAD * DMODEL);               // qk -> qh -> khp -> gate_a
  float* BIG2 = af((size_t)T_PAD * DMODEL);               // vv -> o_pc -> vhp -> gate_b
  float* BIG3 = af((size_t)NHEADS * NHASH * T_PAD * DH);  // soh -> pre-LN scratch
  float* BIG4 = af((size_t)T_PAD * DMODEL);               // att

  // ---- phase 1: encoders ----
  for (int i = 0; i < 6; ++i)
    k_snn<<<1, 256, 0, stream>>>(xo[i], OSZ[i], sW1[i], sB1[i], sW2[i], sB2[i],
                                 hob + i * DMODEL);
  launch_gemm(stream, x_path, NPATH, NPATH, 1024, wsi_W, wsi_b, hp, 1);  // relu

  // ---- phase 2: LSH self-attention over padded WSI tokens ----
  launch_gemm(stream, hp, NPATH, T_PAD, DMODEL, lsh_Wqk, nullptr, BIG1, 0);  // qk
  launch_gemm(stream, hp, NPATH, T_PAD, DMODEL, lsh_Wv, nullptr, BIG2, 0);   // v
  k_lsh_buckets<<<(NHEADS * T_PAD + 255) / 256, 256, 0, stream>>>(BIG1, rot, buckets);
  k_hist<<<NHEADS, 256, 0, stream>>>(buckets, bases);
  k_scatter<<<(NHEADS * NBKT) / 8, 256, 0, stream>>>(buckets, bases, sticker);
  k_lsh_chunk<<<NHEADS * NCHUNK, 256, 0, stream>>>(BIG1, BIG2, sticker, BIG3, slog);
  k_lsh_combine<<<(NHEADS * T_PAD * DH + 255) / 256, 256, 0, stream>>>(BIG3, slog, BIG4);
  launch_gemm(stream, BIG4, NPATH, NPATH, DMODEL, lsh_Wo, lsh_bo, BIG3, 0);  // hs_pre
  k_add_ln<<<NPATH, 256, 0, stream>>>(BIG3, hp, ln_ps.g, ln_ps.b, hs);

  // ---- phase 3: path<-omic cross attention (mha_pc) ----
  launch_gemm(stream, hp, NPATH, NPATH, DMODEL, mha_pc.Wq, mha_pc.bq, BIG1, 0);
  launch_gemm(stream, hob, 6, 6, DMODEL, mha_pc.Wk, mha_pc.bk, k6, 0);
  launch_gemm(stream, hob, 6, 6, DMODEL, mha_pc.Wv, mha_pc.bv, v6, 0);
  k_attn<<<dim3(NPATH, NHEADS), 64, 0, stream>>>(BIG1, k6, v6, BIG2, 6);
  launch_gemm(stream, BIG2, NPATH, NPATH, DMODEL, mha_pc.Wo, mha_pc.bo, BIG3, 0);
  k_add_ln<<<NPATH, 256, 0, stream>>>(BIG3, hp, ln_pc.g, ln_pc.b, hc);

  // ---- phase 4: omic self attention (mha_os) ----
  launch_gemm(stream, hob, 6, 6, DMODEL, mha_os.Wq, mha_os.bq, q6, 0);
  launch_gemm(stream, hob, 6, 6, DMODEL, mha_os.Wk, mha_os.bk, k6, 0);
  launch_gemm(stream, hob, 6, 6, DMODEL, mha_os.Wv, mha_os.bv, v6, 0);
  k_attn<<<dim3(6, NHEADS), 64, 0, stream>>>(q6, k6, v6, o6, 6);
  launch_gemm(stream, o6, 6, 6, DMODEL, mha_os.Wo, mha_os.bo, pre6, 0);
  k_add_ln<<<6, 256, 0, stream>>>(pre6, hob, ln_os.g, ln_os.b, hs2);

  // ---- phase 5: omic<-path cross attention (mha_oc) ----
  launch_gemm(stream, hob, 6, 6, DMODEL, mha_oc.Wq, mha_oc.bq, q6, 0);
  launch_gemm(stream, hp, NPATH, NPATH, DMODEL, mha_oc.Wk, mha_oc.bk, BIG1, 0);
  launch_gemm(stream, hp, NPATH, NPATH, DMODEL, mha_oc.Wv, mha_oc.bv, BIG2, 0);
  k_attn<<<dim3(6, NHEADS), 64, 0, stream>>>(q6, BIG1, BIG2, o6, NPATH);
  launch_gemm(stream, o6, 6, 6, DMODEL, mha_oc.Wo, mha_oc.bo, pre6, 0);
  k_add_ln<<<6, 256, 0, stream>>>(pre6, hob, ln_oc.g, ln_oc.b, hc2);

  // ---- phase 6: gated attention pooling + rho for the four branches ----
  auto gate_pool = [&](const float* x, int N, const GateP& g, const float* rW,
                       const float* rb, float* hout) {
    launch_gemm(stream, x, N, N, DMODEL, g.aW, g.ab, BIG1, 2);  // tanh
    launch_gemm(stream, x, N, N, DMODEL, g.bW, g.bb, BIG2, 3);  // sigmoid
    k_gate_score<<<(N + 255) / 256, 256, 0, stream>>>(BIG1, BIG2, g.cW, g.cb,
                                                      Ascore, N);
    k_softmax_1d<<<1, 256, 0, stream>>>(Ascore, gw, N);
    k_weighted_sum<<<256, 256, 0, stream>>>(x, gw, N, pool_tmp);
    k_vec_linear<1><<<1, 256, 0, stream>>>(pool_tmp, rW, rb, hout);  // relu
  };
  gate_pool(hs, NPATH, g_ps, rps_W, rps_b, hvec + 0 * DMODEL);   // hps
  gate_pool(hc, NPATH, g_pc, rpc_W, rpc_b, hvec + 1 * DMODEL);   // hpc
  gate_pool(hs2, 6, g_os, ros_W, ros_b, hvec + 2 * DMODEL);      // hos
  gate_pool(hc2, 6, g_oc, roc_W, roc_b, hvec + 3 * DMODEL);      // hoc

  // ---- phase 7: fusion + classifiers -> 20 outputs ----
  k_final<<<1, 256, 0, stream>>>(hvec, fq_W, fq_b, fk_W, fk_b, c_W, c_b, cps_W,
                                 cps_b, cpc_W, cpc_b, cos_W, cos_b, coc_W,
                                 coc_b, out);
}
